// LSTMModel_26173530702033
// MI455X (gfx1250) — compile-verified, matrix operand order checked by reading, not run on hardware
//
#include <hip/hip_runtime.h>
#include <hip/hip_bf16.h>

// ---------------------------------------------------------------------------
// 2-layer LSTM (B=128, T=1024, D_IN=256, H=512) + FC(512->128) for gfx1250.
// bf16 WMMA (v_wmma_f32_16x16x32_bf16) with fp32 accumulation.
// Per (t, layer) fused kernel: gate GEMM + LSTM cell in registers.
// Block-shared weight tiles are staged via GLOBAL_LOAD_ASYNC_TO_LDS_B128
// (ASYNCcnt, double-buffered) and consumed by all 8 waves via ds_load_b128,
// removing the 8x redundant global weight stream.
// ---------------------------------------------------------------------------

typedef __bf16 bf16_t;
typedef __attribute__((ext_vector_type(8)))  __bf16 v8bf;
typedef __attribute__((ext_vector_type(16))) __bf16 v16bf;
typedef __attribute__((ext_vector_type(8)))  float  v8f;
typedef __attribute__((ext_vector_type(4)))  float  v4f;

#define B_    128
#define T_    1024
#define DIN_  256
#define H_    512
#define DOUT_ 128
#define BH_   (B_ * H_)

// ---------------------------------------------------------------------------
// math helpers (robust at +-inf)
// ---------------------------------------------------------------------------
__device__ __forceinline__ float fast_sigmoid(float x) {
    return 1.0f / (1.0f + __expf(-x));
}
__device__ __forceinline__ float fast_tanh(float x) {
    float e = __expf(2.0f * x);
    return 1.0f - 2.0f / (e + 1.0f);
}

// ---------------------------------------------------------------------------
// CDNA5 async copy global -> LDS (16B per lane), tracked by ASYNCcnt.
// GVS addressing: mem = SGPR64 + VGPR32 ; LDS = VGPR (byte offset).
// ---------------------------------------------------------------------------
__device__ __forceinline__ void async_b128_to_lds(unsigned lds_off, unsigned goff,
                                                  const void* sbase) {
    asm volatile("global_load_async_to_lds_b128 %0, %1, %2"
                 :
                 : "v"(lds_off), "v"(goff), "s"(sbase)
                 : "memory");
}

template <int N>
__device__ __forceinline__ void wait_asynccnt() {
#if __has_builtin(__builtin_amdgcn_s_wait_asynccnt)
    __builtin_amdgcn_s_wait_asynccnt(N);
#else
    if constexpr (N == 0) asm volatile("s_wait_asynccnt 0x0" ::: "memory");
    else                  asm volatile("s_wait_asynccnt 0x4" ::: "memory");
#endif
}

// low 32 bits of a flat shared address == LDS byte offset
__device__ __forceinline__ unsigned lds_byte_off(const void* p) {
    return (unsigned)(unsigned long long)p;
}

// ---------------------------------------------------------------------------
// WMMA fragment loaders (wave32).
// A 16x32 bf16 layout (ISA 7.12.2): lanes 0-15 -> M=lane, halfs[0..7]=K0..7,
// halfs[8..15]=K16..23 ; lanes 16-31 -> M=lane-16, K8..15 / K24..31.
// B 32x16 mirrored with N = lane&15; works for any row-major source (global
// weights, LDS tiles, bf16 h buffers) via `ld` in elements.
// ---------------------------------------------------------------------------
__device__ __forceinline__ v16bf load_frag_bf16(const bf16_t* __restrict__ base,
                                                int ld, int row0, int k0, int lane) {
    const int half_id = lane >> 4;
    const int r = row0 + (lane & 15);
    const bf16_t* p = base + (size_t)r * ld + k0 + half_id * 8;
    v8bf lo = *(const v8bf*)(p);        // K k0+{0..7} or k0+{8..15}
    v8bf hi = *(const v8bf*)(p + 16);   // K k0+{16..23} or k0+{24..31}
    return __builtin_shufflevector(lo, hi, 0, 1, 2, 3, 4, 5, 6, 7,
                                           8, 9, 10, 11, 12, 13, 14, 15);
}

// Same fragment but sourced from fp32 (layer-0 x), converted in-register.
__device__ __forceinline__ v16bf load_frag_f32cvt(const float* __restrict__ base,
                                                  int ld, int row0, int k0, int lane) {
    const int half_id = lane >> 4;
    const int r = row0 + (lane & 15);
    const float* p = base + (size_t)r * ld + k0 + half_id * 8;
    v4f a0 = *(const v4f*)(p);
    v4f a1 = *(const v4f*)(p + 4);
    v4f b0 = *(const v4f*)(p + 16);
    v4f b1 = *(const v4f*)(p + 20);
    v16bf out;
#pragma unroll
    for (int i = 0; i < 4; ++i) {
        out[i]      = (bf16_t)a0[i];
        out[4 + i]  = (bf16_t)a1[i];
        out[8 + i]  = (bf16_t)b0[i];
        out[12 + i] = (bf16_t)b1[i];
    }
    return out;
}

__device__ __forceinline__ v8f wmma_bf16(v16bf a, v16bf b, v8f c) {
    // (neg_a, A, neg_b, B, c_mod, C, reuse_a, reuse_b)
    return __builtin_amdgcn_wmma_f32_16x16x32_bf16(false, a, false, b,
                                                   (short)0, c, false, false);
}

// ---------------------------------------------------------------------------
// Fused LSTM step.
// Grid: H/16 = 32 blocks (block owns 16 hidden units -> gate rows
// {u..u+15, +512, +1024, +1536} = 64 weight rows). Block: 256 threads =
// 8 waves; wave owns 16 batch rows, keeps 4 gate accumulators in registers.
// Weight rows are streamed through LDS in double-buffered K-chunks of 128
// via async global->LDS copies.
// ---------------------------------------------------------------------------
template <int DIN, bool IN_F32>
__global__ __launch_bounds__(256) void lstm_step_kernel(
    const float*  __restrict__ xf,     // layer0: x + t*DIN, ld = xld
    int xld,
    const bf16_t* __restrict__ xb,     // layer1: h of layer0 at t, ld = H
    const bf16_t* __restrict__ Wih,    // [4H, DIN] bf16 row-major
    const bf16_t* __restrict__ Whh,    // [4H, H]   bf16 row-major
    const float*  __restrict__ b_ih,   // [4H]
    const float*  __restrict__ b_hh,   // [4H]
    const bf16_t* __restrict__ h_prev, // [B, H]
    bf16_t*       __restrict__ h_out,  // [B, H]
    float*        __restrict__ c_state)// [B, H] (block-private columns)
{
    constexpr int CK     = 128;            // K-chunk staged per buffer
    constexpr int RS     = CK + 8;         // padded LDS row stride (elements)
    constexpr int KTOT   = DIN + H_;       // unified weight K-stream
    constexpr int NCHUNK = KTOT / CK;      // 6 (layer0) or 8 (layer1)

    __shared__ __align__(16) bf16_t wbuf[2][64 * RS];   // ~34 KB

    const int tid  = threadIdx.x;
    const int lane = tid & 31;
    const int wave = tid >> 5;            // 0..7
    const int row0 = wave * 16;           // batch-tile origin
    const int u0   = blockIdx.x * 16;     // hidden-unit-tile origin

    // stage one K-chunk (64 rows x 128 K of the unified Wih|Whh stream) into
    // wbuf[buf]; 4 async b128 ops per thread (1024 x 16B = 16 KB payload).
    auto stage = [&](int ch, int buf) {
        const int kg = ch * CK;
        const bf16_t* src;
        int ld, kb;
        if (kg < DIN) { src = Wih; ld = DIN; kb = kg; }
        else          { src = Whh; ld = H_;  kb = kg - DIN; }
        const unsigned lbase = lds_byte_off(&wbuf[buf][0]);
#pragma unroll
        for (int j = 0; j < 4; ++j) {
            const int c  = j * 256 + tid;     // 16B-chunk id, coalesced
            const int lr = c >> 4;            // local weight row 0..63
            const int cc = c & 15;            // 16B chunk within row
            const int grow = (lr >> 4) * H_ + u0 + (lr & 15);
            const unsigned goff = (unsigned)(grow * ld + kb) * 2u + (unsigned)cc * 16u;
            const unsigned loff = lbase + (unsigned)lr * (RS * 2) + (unsigned)cc * 16u;
            async_b128_to_lds(loff, goff, src);
        }
    };

    v8f acc[4] = {};                      // i, f, g, o accumulators (16x16 f32)

    stage(0, 0);
    stage(1, 1);

    for (int ch = 0; ch < NCHUNK; ++ch) {
        if (ch == NCHUNK - 1) wait_asynccnt<0>();  // last chunk: drain
        else                  wait_asynccnt<4>();  // keep next chunk in flight
        __syncthreads();                           // all waves' copies visible

        const bf16_t* wb = &wbuf[ch & 1][0];
#pragma unroll
        for (int kk = 0; kk < CK; kk += 32) {
            const int kg = ch * CK + kk;
            v16bf a;
            if (kg < DIN) {
                if (IN_F32) a = load_frag_f32cvt(xf, xld, row0, kg, lane);
                else        a = load_frag_bf16(xb, H_, row0, kg, lane);
            } else {
                a = load_frag_bf16(h_prev, H_, row0, kg - DIN, lane);
            }
#pragma unroll
            for (int g = 0; g < 4; ++g) {
                v16bf b = load_frag_bf16(wb, RS, g * 16, kk, lane); // ds_load
                acc[g] = wmma_bf16(a, b, acc[g]);   // A reused across 4 gates
            }
        }
        __syncthreads();                           // buffer free to overwrite
        if (ch + 2 < NCHUNK) stage(ch + 2, ch & 1);
    }

    // Accumulator element (VGPR r, lane l) = C[M = r + 8*(l>>4), N = l&15].
    const int n   = lane & 15;
    const int col = u0 + n;
    const float bi = b_ih[0 * H_ + col] + b_hh[0 * H_ + col];
    const float bf = b_ih[1 * H_ + col] + b_hh[1 * H_ + col];
    const float bg = b_ih[2 * H_ + col] + b_hh[2 * H_ + col];
    const float bo = b_ih[3 * H_ + col] + b_hh[3 * H_ + col];

#pragma unroll
    for (int r = 0; r < 8; ++r) {
        const int m = r + 8 * (lane >> 4);
        const size_t idx = (size_t)(row0 + m) * H_ + col;
        const float ig = fast_sigmoid(acc[0][r] + bi);
        const float fg = fast_sigmoid(acc[1][r] + bf);
        const float gg = fast_tanh(acc[2][r] + bg);
        const float og = fast_sigmoid(acc[3][r] + bo);
        const float cn = fg * c_state[idx] + ig * gg;
        c_state[idx] = cn;
        h_out[idx]   = (bf16_t)(og * fast_tanh(cn));
    }
}

// ---------------------------------------------------------------------------
// Final FC: out[128,128] = h[128,512] @ fc_w^T + fc_b  (WMMA, 64 tiles)
// ---------------------------------------------------------------------------
__global__ __launch_bounds__(256) void fc_kernel(
    const bf16_t* __restrict__ h,    // [B, H]
    const bf16_t* __restrict__ w,    // [DOUT, H] row-major
    const float*  __restrict__ bias, // [DOUT]
    float*        __restrict__ out)  // [B, DOUT]
{
    const int lane = threadIdx.x & 31;
    const int wave = threadIdx.x >> 5;
    const int row0 = wave * 16;
    const int col0 = blockIdx.x * 16;

    v8f acc = {};
    for (int k = 0; k < H_; k += 32) {
        v16bf a = load_frag_bf16(h, H_, row0, k, lane);
        v16bf b = load_frag_bf16(w, H_, col0, k, lane);
        acc = wmma_bf16(a, b, acc);
    }
    const int n = lane & 15;
    const float bb = bias[col0 + n];
#pragma unroll
    for (int r = 0; r < 8; ++r) {
        const int m = r + 8 * (lane >> 4);
        out[(size_t)(row0 + m) * DOUT_ + col0 + n] = acc[r] + bb;
    }
}

// ---------------------------------------------------------------------------
// One-time helpers: fp32 -> bf16 weight conversion, state zero-init
// ---------------------------------------------------------------------------
__global__ void cvt_f32_to_bf16(const float* __restrict__ in,
                                bf16_t* __restrict__ outp, int n) {
    int i = blockIdx.x * blockDim.x + threadIdx.x;
    if (i < n) outp[i] = (bf16_t)in[i];
}

__global__ void zero_state_kernel(float* __restrict__ c0, float* __restrict__ c1,
                                  bf16_t* __restrict__ h0, bf16_t* __restrict__ h1,
                                  int n) {
    int i = blockIdx.x * blockDim.x + threadIdx.x;
    if (i < n) {
        c0[i] = 0.0f;
        c1[i] = 0.0f;
        h0[i] = (bf16_t)0.0f;   // ping buffer (index 0) read at t=0
        h1[i] = (bf16_t)0.0f;
    }
}

// ---------------------------------------------------------------------------
// Host launcher
// ---------------------------------------------------------------------------
extern "C" void kernel_launch(void* const* d_in, const int* in_sizes, int n_in,
                              void* d_out, int out_size, void* d_ws, size_t ws_size,
                              hipStream_t stream) {
    (void)in_sizes; (void)n_in; (void)out_size; (void)ws_size;

    const float* x    = (const float*)d_in[0];   // [B, T, D_IN]
    const float* Wih0 = (const float*)d_in[1];   // [4H, D_IN]
    const float* Whh0 = (const float*)d_in[2];   // [4H, H]
    const float* bih0 = (const float*)d_in[3];
    const float* bhh0 = (const float*)d_in[4];
    const float* Wih1 = (const float*)d_in[5];   // [4H, H]
    const float* Whh1 = (const float*)d_in[6];   // [4H, H]
    const float* bih1 = (const float*)d_in[7];
    const float* bhh1 = (const float*)d_in[8];
    const float* fcw  = (const float*)d_in[9];   // [DOUT, H]
    const float* fcb  = (const float*)d_in[10];
    float* out = (float*)d_out;

    // workspace carve-up (~8.5 MB, 256B aligned)
    char* ws = (char*)d_ws;
    size_t off = 0;
    auto carve = [&](size_t bytes) -> char* {
        char* p = ws + off;
        off = (off + bytes + 255) & ~(size_t)255;
        return p;
    };
    bf16_t* Wih0b = (bf16_t*)carve((size_t)4 * H_ * DIN_ * sizeof(bf16_t));
    bf16_t* Whh0b = (bf16_t*)carve((size_t)4 * H_ * H_   * sizeof(bf16_t));
    bf16_t* Wih1b = (bf16_t*)carve((size_t)4 * H_ * H_   * sizeof(bf16_t));
    bf16_t* Whh1b = (bf16_t*)carve((size_t)4 * H_ * H_   * sizeof(bf16_t));
    bf16_t* fcwb  = (bf16_t*)carve((size_t)DOUT_ * H_    * sizeof(bf16_t));
    bf16_t* h0buf = (bf16_t*)carve((size_t)2 * BH_ * sizeof(bf16_t)); // ping/pong
    bf16_t* h1buf = (bf16_t*)carve((size_t)2 * BH_ * sizeof(bf16_t)); // ping/pong
    float*  c0    = (float*)carve((size_t)BH_ * sizeof(float));
    float*  c1    = (float*)carve((size_t)BH_ * sizeof(float));

    auto cvt = [&](const float* src, bf16_t* dst, int n) {
        cvt_f32_to_bf16<<<(n + 255) / 256, 256, 0, stream>>>(src, dst, n);
    };
    cvt(Wih0, Wih0b, 4 * H_ * DIN_);
    cvt(Whh0, Whh0b, 4 * H_ * H_);
    cvt(Wih1, Wih1b, 4 * H_ * H_);
    cvt(Whh1, Whh1b, 4 * H_ * H_);
    cvt(fcw,  fcwb,  DOUT_ * H_);
    zero_state_kernel<<<(BH_ + 255) / 256, 256, 0, stream>>>(c0, c1, h0buf, h1buf, BH_);

    // sequential scan over T: two fused step kernels per timestep
    for (int t = 0; t < T_; ++t) {
        const int pi = t & 1;          // read h index
        const int po = (t + 1) & 1;    // write h index
        lstm_step_kernel<DIN_, true><<<H_ / 16, 256, 0, stream>>>(
            x + (size_t)t * DIN_, T_ * DIN_, (const bf16_t*)nullptr,
            Wih0b, Whh0b, bih0, bhh0,
            h0buf + (size_t)pi * BH_, h0buf + (size_t)po * BH_, c0);
        lstm_step_kernel<H_, false><<<H_ / 16, 256, 0, stream>>>(
            (const float*)nullptr, 0, h0buf + (size_t)po * BH_,
            Wih1b, Whh1b, bih1, bhh1,
            h1buf + (size_t)pi * BH_, h1buf + (size_t)po * BH_, c1);
    }

    // after t = T-1, final h of layer 1 sits at index (T & 1) == 0
    fc_kernel<<<DOUT_ / 16, 256, 0, stream>>>(h1buf, fcwb, fcb, out);
}
